// GCNConv_18957985644925
// MI455X (gfx1250) — matrix operand order repaired and checked by reading, not compiled, not run on hardware
//
#include <hip/hip_runtime.h>
#include <math.h>

// Problem constants (match reference)
#define NN   100000
#define EE   640000
#define DD   128
#define HIDD 128
#define OUTD 128
#define RR   8

typedef float v8f  __attribute__((ext_vector_type(8)));
typedef __bf16 v16bf __attribute__((ext_vector_type(16)));

union FragU {
    uint4 u[2];
    v16bf v;
};

__device__ __forceinline__ unsigned short f2bf(float f) {
    unsigned int u = __float_as_uint(f);
    unsigned int r = u + 0x7fffu + ((u >> 16) & 1u);   // round-to-nearest-even
    return (unsigned short)(r >> 16);
}

// ---------------------------------------------------------------------------
// Zero a float region (grid-stride, float4 stores)
// ---------------------------------------------------------------------------
__global__ void zero_f32(float4* __restrict__ p, long n4) {
    long i = (long)blockIdx.x * blockDim.x + threadIdx.x;
    long stride = (long)gridDim.x * blockDim.x;
    float4 z = make_float4(0.f, 0.f, 0.f, 0.f);
    for (; i < n4; i += stride) p[i] = z;
}

// ---------------------------------------------------------------------------
// Repack a row-major f32 weight matrix [K, No] into bf16 WMMA-B-fragment-
// linear order:  dst[((ksIdx*NT + nt)*32 + lane)*16 + i]  holds element
// (k = ksIdx*32 + (lane>>4)*16 + i ,  n = nt*16 + (lane&15)).
// ksIdx-major => a [R*128, No] pack is also 8 contiguous per-relation packs.
// ---------------------------------------------------------------------------
__global__ void pack_b(const float* __restrict__ W, unsigned short* __restrict__ dst,
                       int K, int No) {
    int NT = No >> 4;
    long total = (long)K * No;
    long p = (long)blockIdx.x * blockDim.x + threadIdx.x;
    long stride = (long)gridDim.x * blockDim.x;
    for (; p < total; p += stride) {
        int i    = (int)(p & 15);
        int lane = (int)((p >> 4) & 31);
        long rest = p >> 9;
        int nt    = (int)(rest % NT);
        long ksIdx = rest / NT;
        int k = (int)(ksIdx * 32) + ((lane >> 4) << 4) + i;
        int n = (nt << 4) + (lane & 15);
        dst[p] = f2bf(W[(long)k * No + n]);
    }
}

// ---------------------------------------------------------------------------
// Per-relation edge scatter:  for edges with etype==r:  agg[dst, :] += x[src, :]
// One wave32 per edge; lane l covers features l, l+32, l+64, l+96 so loads and
// f32 atomics are fully coalesced.  agg is 51 MB -> L2-resident on MI455X.
// ---------------------------------------------------------------------------
__global__ __launch_bounds__(256) void scatter_rel(
    const float* __restrict__ x, const int* __restrict__ src,
    const int* __restrict__ dst, const int* __restrict__ et,
    float* __restrict__ agg, int E, int r) {
    int gid  = blockIdx.x * blockDim.x + threadIdx.x;
    int lane = gid & 31;
    int wave = gid >> 5;
    int nwaves = (gridDim.x * blockDim.x) >> 5;
    for (int e = wave; e < E; e += nwaves) {
        if (et[e] != r) continue;
        int s = src[e], d = dst[e];
        const float* xs = x + (long)s * DD;
        float* o = agg + (long)d * DD;
        #pragma unroll
        for (int j = 0; j < 4; ++j) {
            atomicAdd(&o[lane + 32 * j], xs[lane + 32 * j]);
        }
    }
}

// ---------------------------------------------------------------------------
// Fused GEMM: out[m,:] {=, +=} act( [A1[m,:K1] | A2[m,:K2]] @ Bpacked (+bias) )
//   A1/A2 : f32 row-major, logically concatenated along K (K1,K2 mult. of 32)
//   Bpacked: bf16 fragment-linear (see pack_b), K = K1+K2 rows, No cols
//   acc=1 -> accumulate into out;  act=1 -> tanh;  bias may be null
// Block = 16 output rows, 256 threads (8 waves).  A tile staged into LDS as
// bf16 (row-major, read in contiguous k = 16h + s order); one wmma per 32-wide K step.
// ---------------------------------------------------------------------------
#define LDS_KMAX 384
#define LDS_LDK  (LDS_KMAX + 8)

__global__ __launch_bounds__(256) void gemm_fused(
    const float* __restrict__ A1, int K1,
    const float* __restrict__ A2, int K2,
    const unsigned short* __restrict__ PB,
    const float* __restrict__ bias,
    float* __restrict__ out, int No, int N, int act, int acc) {

    __shared__ __align__(16) unsigned short ldsA[16][LDS_LDK];

    const int K   = K1 + K2;
    const int m0  = blockIdx.x * 16;
    const int tid = threadIdx.x;

    // Stage 16 x K A-tile into LDS (f32 -> bf16), coalesced along K.
    for (int row = 0; row < 16; ++row) {
        int m = m0 + row;
        const float* a1 = A1 + (long)m * K1;
        const float* a2 = A2 + (long)m * K2;
        for (int k = tid; k < K; k += 256) {
            float v = 0.f;
            if (m < N) v = (k < K1) ? a1[k] : a2[k - K1];
            ldsA[row][k] = f2bf(v);
        }
    }
    __syncthreads();

    const int lane = tid & 31;
    const int wave = tid >> 5;
    const int hl   = lane >> 4;   // half-wave select
    const int rc   = lane & 15;   // row (A) / col (B,C) within tile
    const int NT   = No >> 4;

    for (int nt = wave; nt < NT; nt += 8) {
        v8f c = {};
        for (int ks = 0; ks < K; ks += 32) {
            // A fragment: lane<16 -> row rc, K = ks+{0..15};
            //             lane>=16 -> row rc, K = ks+{16..31}.
            FragU a;
            a.u[0] = *(const uint4*)&ldsA[rc][ks + hl * 16];
            a.u[1] = *(const uint4*)&ldsA[rc][ks + hl * 16 + 8];
            // B fragment: 16 contiguous bf16 from the packed buffer.
            FragU b;
            const unsigned short* pb =
                PB + ((((long)(ks >> 5)) * NT + nt) * 32 + lane) * 16;
            b.u[0] = *(const uint4*)(pb);
            b.u[1] = *(const uint4*)(pb + 8);
            c = __builtin_amdgcn_wmma_f32_16x16x32_bf16(
                    false, a.v, false, b.v, (short)0, c, false, false);
        }
        // Epilogue: C layout -> lane holds col rc; VGPR j is row j + 8*hl.
        int colg = (nt << 4) + rc;
        float bv = bias ? bias[colg] : 0.f;
        #pragma unroll
        for (int j = 0; j < 8; ++j) {
            int m = m0 + j + hl * 8;
            if (m < N) {
                long oidx = (long)m * No + colg;
                float v = c[j] + bv;
                if (acc) v += out[oidx];
                if (act) v = tanhf(v);
                out[oidx] = v;
            }
        }
    }
}

// ---------------------------------------------------------------------------
// Host-side orchestration
// ---------------------------------------------------------------------------
extern "C" void kernel_launch(void* const* d_in, const int* in_sizes, int n_in,
                              void* d_out, int out_size, void* d_ws, size_t ws_size,
                              hipStream_t stream) {
    const float* x        = (const float*)d_in[0];
    const int*   src      = (const int*)d_in[1];
    const int*   dst      = (const int*)d_in[2];
    const int*   etype    = (const int*)d_in[3];
    const float* W_rel    = (const float*)d_in[4];   // [R, D, OUT] == [1024,128]
    const float* loop_w   = (const float*)d_in[5];   // [128,128]
    const float* rel_bias = (const float*)d_in[6];   // [128]
    const float* W1       = (const float*)d_in[7];   // [256,256]
    const float* b1       = (const float*)d_in[8];   // [256]
    const float* W2       = (const float*)d_in[9];   // [384,128]
    const float* b2       = (const float*)d_in[10];  // [128]
    float* out = (float*)d_out;

    const int N = NN, E = EE;

    // Workspace layout (all offsets 256B aligned); total ~205 MB.
    char* ws = (char*)d_ws;
    size_t off = 0;
    auto alloc = [&](size_t bytes) {
        size_t o = off;
        off += (bytes + 255) & ~(size_t)255;
        return ws + o;
    };
    float* agg = (float*)alloc((size_t)N * DD * sizeof(float));        //  51.2 MB
    float* msg = (float*)alloc((size_t)N * OUTD * sizeof(float));      //  51.2 MB
    float* mid = (float*)alloc((size_t)N * 2 * HIDD * sizeof(float));  // 102.4 MB
    unsigned short* pb_rel  = (unsigned short*)alloc((size_t)RR * DD * OUTD * 2);
    unsigned short* pb_loop = (unsigned short*)alloc((size_t)DD * OUTD * 2);
    unsigned short* pb_w1   = (unsigned short*)alloc((size_t)(DD + HIDD) * 2 * HIDD * 2);
    unsigned short* pb_w2   = (unsigned short*)alloc((size_t)(DD + 2 * HIDD) * OUTD * 2);

    // 1) repack weights to bf16 fragment order
    {
        int t;
        t = RR * DD * OUTD;          // W_rel as [1024,128] -> 8 contiguous packs
        pack_b<<<(t + 255) / 256, 256, 0, stream>>>(W_rel, pb_rel, RR * DD, OUTD);
        t = DD * OUTD;
        pack_b<<<(t + 255) / 256, 256, 0, stream>>>(loop_w, pb_loop, DD, OUTD);
        t = (DD + HIDD) * 2 * HIDD;
        pack_b<<<(t + 255) / 256, 256, 0, stream>>>(W1, pb_w1, DD + HIDD, 2 * HIDD);
        t = (DD + 2 * HIDD) * OUTD;
        pack_b<<<(t + 255) / 256, 256, 0, stream>>>(W2, pb_w2, DD + 2 * HIDD, OUTD);
    }

    const int nblk = (N + 15) / 16;

    // 2) msg = x @ loop_w + rel_bias   (initializes msg)
    gemm_fused<<<nblk, 256, 0, stream>>>(x, DD, x, 0,
                                         pb_loop, rel_bias, msg, OUTD, N,
                                         /*act=*/0, /*acc=*/0);

    // 3) per-relation L2-resident passes: agg<-0; agg[dst]+=x[src]; msg += agg@W_rel[r]
    {
        long n4 = (long)N * DD / 4;
        for (int r = 0; r < RR; ++r) {
            zero_f32<<<2048, 256, 0, stream>>>((float4*)agg, n4);
            scatter_rel<<<E / 8, 256, 0, stream>>>(x, src, dst, etype, agg, E, r);
            gemm_fused<<<nblk, 256, 0, stream>>>(agg, DD, x, 0,
                                                 pb_rel + (size_t)r * DD * OUTD,
                                                 nullptr, msg, OUTD, N,
                                                 /*act=*/0, /*acc=*/1);
        }
    }

    // 4) mid = tanh([x | msg] @ W1 + b1)
    gemm_fused<<<nblk, 256, 0, stream>>>(x, DD, msg, HIDD,
                                         pb_w1, b1, mid, 2 * HIDD, N,
                                         /*act=*/1, /*acc=*/0);
    // 5) out = [x | mid] @ W2 + b2
    gemm_fused<<<nblk, 256, 0, stream>>>(x, DD, mid, 2 * HIDD,
                                         pb_w2, b2, out, OUTD, N,
                                         /*act=*/0, /*acc=*/0);
}
